// PMA_batch_74509092651637
// MI455X (gfx1250) — matrix operand order repaired and checked by reading, not compiled, not run on hardware
//
#include <hip/hip_runtime.h>
#include <hip/hip_bf16.h>

#define Bq 8
#define Nn 4096
#define Dd 256
#define Ee 32768
#define Hh 4
#define Cc 64
#define HC 256
#define NEG_SLOPE 0.2f
#define CHUNK 64

typedef __attribute__((ext_vector_type(16))) __bf16 bf16x16;
typedef __attribute__((ext_vector_type(8)))  __bf16 bf16x8;
typedef __attribute__((ext_vector_type(8)))  float  f32x8;

// ---------- conversion f32 -> bf16 ----------
__global__ void k_f32_to_bf16(const float* __restrict__ in, __bf16* __restrict__ out, int n) {
    int i = blockIdx.x * blockDim.x + threadIdx.x;
    if (i < n) out[i] = (__bf16)in[i];
}

// ---------- wA[h,d] = sum_c att_r[h,c] * W_K[h*C+c, d]; ab[h] = sum_c att_r*b_K ----------
__global__ void k_watt(const float* __restrict__ W_K, const float* __restrict__ b_K,
                       const float* __restrict__ att_r, float* __restrict__ wA,
                       float* __restrict__ ab) {
    int idx = blockIdx.x * blockDim.x + threadIdx.x;   // h*Dd + d
    if (idx >= Hh * Dd) return;
    int h = idx / Dd, d = idx % Dd;
    float s = 0.f;
    for (int c = 0; c < Cc; ++c)
        s += att_r[h * Cc + c] * W_K[(size_t)(h * Cc + c) * Dd + d];
    wA[idx] = s;
    if (d == 0) {
        float sb = 0.f;
        for (int c = 0; c < Cc; ++c) sb += att_r[h * Cc + c] * b_K[h * Cc + c];
        ab[h] = sb;
    }
}

// ---------- alpha[b,n,h] = x[b,n,:] . wA[h,:] + ab[h] ----------
__global__ void k_alpha(const float* __restrict__ x, const float* __restrict__ wA,
                        const float* __restrict__ ab, float* __restrict__ alpha) {
    int idx = blockIdx.x * blockDim.x + threadIdx.x;   // (b*Nn+n)*Hh + h
    if (idx >= Bq * Nn * Hh) return;
    int h = idx & (Hh - 1);
    int row = idx >> 2;
    const float4* xr = (const float4*)(x + (size_t)row * Dd);
    const float4* wr = (const float4*)(wA + h * Dd);
    float s = 0.f;
    #pragma unroll 8
    for (int i = 0; i < Dd / 4; ++i) {
        float4 a = xr[i], w = wr[i];
        s += a.x * w.x + a.y * w.y + a.z * w.z + a.w * w.w;
    }
    alpha[idx] = s + ab[h];
}

// ---------- CSR build: count / scan / scatter / per-bucket sort ----------
__global__ void k_count(const int* __restrict__ dst, int* __restrict__ deg) {
    int e = blockIdx.x * blockDim.x + threadIdx.x;
    if (e < Ee) atomicAdd(&deg[dst[e]], 1);
}

__global__ __launch_bounds__(1024) void k_scan(const int* __restrict__ deg,
                                               int* __restrict__ rowstart,
                                               int* __restrict__ cursor) {
    __shared__ int sh[1024];
    int t = threadIdx.x;                       // one block; 4 elems/thread, Nn = 4096
    int d0 = deg[4 * t + 0], d1 = deg[4 * t + 1], d2 = deg[4 * t + 2], d3 = deg[4 * t + 3];
    int local = d0 + d1 + d2 + d3;
    sh[t] = local;
    __syncthreads();
    for (int off = 1; off < 1024; off <<= 1) { // Hillis-Steele inclusive scan
        int v = (t >= off) ? sh[t - off] : 0;
        __syncthreads();
        sh[t] += v;
        __syncthreads();
    }
    int excl = sh[t] - local;
    rowstart[4 * t + 0] = excl;
    rowstart[4 * t + 1] = excl + d0;
    rowstart[4 * t + 2] = excl + d0 + d1;
    rowstart[4 * t + 3] = excl + d0 + d1 + d2;
    cursor[4 * t + 0] = excl;
    cursor[4 * t + 1] = excl + d0;
    cursor[4 * t + 2] = excl + d0 + d1;
    cursor[4 * t + 3] = excl + d0 + d1 + d2;
    if (t == 1023) rowstart[Nn] = sh[1023];
}

__global__ void k_scatter(const int* __restrict__ dst, int* __restrict__ cursor,
                          int* __restrict__ eidx) {
    int e = blockIdx.x * blockDim.x + threadIdx.x;
    if (e < Ee) {
        int pos = atomicAdd(&cursor[dst[e]], 1);
        eidx[pos] = e;
    }
}

// sort each bucket by edge id (determinism) and materialize source ids
__global__ void k_sortbuckets(const int* __restrict__ rowstart, int* __restrict__ eidx,
                              const int* __restrict__ src, int* __restrict__ srcs) {
    int n = blockIdx.x * blockDim.x + threadIdx.x;
    if (n >= Nn) return;
    int a = rowstart[n], b = rowstart[n + 1];
    for (int i = a + 1; i < b; ++i) {          // insertion sort (avg deg ~8)
        int key = eidx[i], j = i - 1;
        while (j >= a && eidx[j] > key) { eidx[j + 1] = eidx[j]; --j; }
        eidx[j + 1] = key;
    }
    for (int i = a; i < b; ++i) srcs[i] = src[eidx[i]];
}

// ---------- bf16 WMMA GEMM: out[M,256] = A[M,256] @ W[256,256]^T (+bias)(+relu) ----------
__global__ __launch_bounds__(128) void k_gemm_bf16(
    const __bf16* __restrict__ A, const __bf16* __restrict__ W,
    const float* __restrict__ bias, float* __restrict__ outF,
    __bf16* __restrict__ outB, int M, int relu)
{
    const int KK = 256;
    int wave = threadIdx.x >> 5;
    int lane = threadIdx.x & 31;
    int m0 = blockIdx.x * 16;
    int n0 = wave * 64;
    int mlane = lane & 15;
    int half = lane >> 4;

    const __bf16* Arow = A + (size_t)(m0 + mlane) * KK;
    bf16x16 a[8];
    #pragma unroll
    for (int kk = 0; kk < 8; ++kk) {
        bf16x8 lo = *(const bf16x8*)(Arow + kk * 32 + half * 8);
        bf16x8 hi = *(const bf16x8*)(Arow + kk * 32 + 16 + half * 8);
        a[kk] = __builtin_shufflevector(lo, hi, 0,1,2,3,4,5,6,7,8,9,10,11,12,13,14,15);
    }

    int kb = half * 8;
    #pragma unroll
    for (int nt = 0; nt < 4; ++nt) {
        int ncol = n0 + nt * 16 + mlane;
        const __bf16* Wrow = W + (size_t)ncol * KK + kb;
        f32x8 acc = {};
        #pragma unroll
        for (int kk = 0; kk < 8; ++kk) {
            bf16x8 blo = *(const bf16x8*)(Wrow + kk * 32);
            bf16x8 bhi = *(const bf16x8*)(Wrow + kk * 32 + 16);
            bf16x16 bm = __builtin_shufflevector(blo, bhi, 0,1,2,3,4,5,6,7,8,9,10,11,12,13,14,15);
            acc = __builtin_amdgcn_wmma_f32_16x16x32_bf16(
                false, a[kk], false, bm, (short)0, acc, false, false);
        }
        float bs = bias ? bias[ncol] : 0.f;
        #pragma unroll
        for (int v = 0; v < 8; ++v) {
            float val = acc[v] + bs;
            if (relu) val = fmaxf(val, 0.f);
            size_t o = (size_t)(m0 + v + half * 8) * HC + ncol;
            if (outF) outF[o] = val;
            if (outB) outB[o] = (__bf16)val;
        }
    }
}

// ---------- fused gather softmax-aggregation + att_r residual + LN0 ----------
// grid = Bq*Nn blocks, 256 threads (one per channel). Gather-based: no float atomics.
__global__ __launch_bounds__(256) void k_aggln(
    const float* __restrict__ alpha, const __bf16* __restrict__ xVb,
    const int* __restrict__ rowstart, const int* __restrict__ srcs,
    const float* __restrict__ attr, const float* __restrict__ g,
    const float* __restrict__ bb, float* __restrict__ out0f,
    __bf16* __restrict__ out0b)
{
    int bn = blockIdx.x;
    int b = bn >> 12;                 // / Nn
    int n = bn & (Nn - 1);
    int t = threadIdx.x;
    int h4 = t & 3;                   // strided (e,h) work keeps h == t&3
    int beg = rowstart[n], end = rowstart[n + 1];
    int deg = end - beg;

    __shared__ float red[256];
    __shared__ float smax[4], sden[4];
    __shared__ int   ssrc[CHUNK];
    __shared__ float sw[CHUNK * 4];

    // pass 1: per-head max of leaky(alpha[src])
    float m = -3.4e38f;
    for (int i = t; i < deg * 4; i += 256) {
        int e = i >> 2, h = i & 3;
        float a = alpha[((size_t)b * Nn + srcs[beg + e]) * Hh + h];
        a = (a >= 0.f) ? a : NEG_SLOPE * a;
        m = fmaxf(m, a);
    }
    red[t] = m;
    __syncthreads();
    for (int s = 128; s >= 4; s >>= 1) {
        if (t < s) red[t] = fmaxf(red[t], red[t + s]);
        __syncthreads();
    }
    if (t < 4) {
        float mm = red[t];
        smax[t] = (mm > -3.0e38f && mm < 3.0e38f) ? mm : 0.f;  // ref: non-finite -> 0
    }
    __syncthreads();

    // pass 2: per-head denominator
    float mh = smax[h4];
    float dsum = 0.f;
    for (int i = t; i < deg * 4; i += 256) {
        int e = i >> 2, h = i & 3;
        float a = alpha[((size_t)b * Nn + srcs[beg + e]) * Hh + h];
        a = (a >= 0.f) ? a : NEG_SLOPE * a;
        dsum += __expf(a - mh);
    }
    red[t] = dsum;
    __syncthreads();
    for (int s = 128; s >= 4; s >>= 1) {
        if (t < s) red[t] += red[t + s];
        __syncthreads();
    }
    if (t < 4) sden[t] = red[t] + 1e-16f;
    __syncthreads();

    // pass 3: chunked weighted gather of xV rows (register accumulation)
    int ch = t;
    int h = ch >> 6;                  // / Cc
    float acc = 0.f;
    for (int cb = 0; cb < deg; cb += CHUNK) {
        int ce = min(CHUNK, deg - cb);
        __syncthreads();
        for (int i = t; i < ce * 4; i += 256) {
            int e = i >> 2, hh = i & 3;
            int s_ = srcs[beg + cb + e];
            if (hh == 0) ssrc[e] = s_;
            float a = alpha[((size_t)b * Nn + s_) * Hh + hh];
            a = (a >= 0.f) ? a : NEG_SLOPE * a;
            sw[e * 4 + hh] = __expf(a - smax[hh]) / sden[hh];
        }
        __syncthreads();
        for (int e = 0; e < ce; ++e)
            acc += (float)xVb[((size_t)b * Nn + ssrc[e]) * HC + ch] * sw[e * 4 + h];
    }

    // residual + LayerNorm 0
    float v = acc + attr[ch];
    __syncthreads();
    red[t] = v;
    __syncthreads();
    for (int s = 128; s > 0; s >>= 1) {
        if (t < s) red[t] += red[t + s];
        __syncthreads();
    }
    float mean = red[0] * (1.f / HC);
    __syncthreads();
    red[t] = v * v;
    __syncthreads();
    for (int s = 128; s > 0; s >>= 1) {
        if (t < s) red[t] += red[t + s];
        __syncthreads();
    }
    float var = red[0] * (1.f / HC) - mean * mean;
    float rs = rsqrtf(var + 1e-5f);
    float y = (v - mean) * rs * g[ch] + bb[ch];
    out0f[(size_t)bn * HC + ch] = y;
    out0b[(size_t)bn * HC + ch] = (__bf16)y;
}

// ---------- final: out = LN(out0 + relu(rff)) ----------
__global__ __launch_bounds__(256) void k_ln1(const float* __restrict__ out0,
                                             const float* __restrict__ rff,
                                             const float* __restrict__ g,
                                             const float* __restrict__ bb,
                                             float* __restrict__ out) {
    int row = blockIdx.x;
    int t = threadIdx.x;
    size_t i = (size_t)row * HC + t;
    float v = out0[i] + fmaxf(rff[i], 0.f);
    __shared__ float s1[256], s2[256];
    s1[t] = v; s2[t] = v * v;
    __syncthreads();
    for (int s = 128; s > 0; s >>= 1) {
        if (t < s) { s1[t] += s1[t + s]; s2[t] += s2[t + s]; }
        __syncthreads();
    }
    float mean = s1[0] * (1.f / HC);
    float var = s2[0] * (1.f / HC) - mean * mean;
    float rs = rsqrtf(var + 1e-5f);
    out[i] = (v - mean) * rs * g[t] + bb[t];
}

extern "C" void kernel_launch(void* const* d_in, const int* in_sizes, int n_in,
                              void* d_out, int out_size, void* d_ws, size_t ws_size,
                              hipStream_t stream) {
    const float* x     = (const float*)d_in[0];
    const int*   ei    = (const int*)d_in[1];
    const float* W_K   = (const float*)d_in[2];
    const float* b_K   = (const float*)d_in[3];
    const float* W_V   = (const float*)d_in[4];
    const float* b_V   = (const float*)d_in[5];
    const float* att_r = (const float*)d_in[6];
    const float* ln0_g = (const float*)d_in[7];
    const float* ln0_b = (const float*)d_in[8];
    const float* ln1_g = (const float*)d_in[9];
    const float* ln1_b = (const float*)d_in[10];
    const float* W1    = (const float*)d_in[11];
    const float* b1    = (const float*)d_in[12];
    const float* W2    = (const float*)d_in[13];
    const float* b2    = (const float*)d_in[14];
    const int* src = ei;
    const int* dst = ei + Ee;
    float* out = (float*)d_out;

    const int ROWS = Bq * Nn;                          // 32768
    char* ws = (char*)d_ws;
    size_t off = 0;
    auto take = [&](size_t bytes) { char* p = ws + off; off += (bytes + 255) & ~(size_t)255; return p; };

    __bf16* x_bf   = (__bf16*)take((size_t)ROWS * Dd * 2);       // 16 MB
    __bf16* wv_bf  = (__bf16*)take((size_t)HC * Dd * 2);
    __bf16* w1_bf  = (__bf16*)take((size_t)HC * HC * 2);
    __bf16* w2_bf  = (__bf16*)take((size_t)HC * HC * 2);
    float*  wA     = (float*)take((size_t)Hh * Dd * 4);
    float*  ab     = (float*)take(256);
    float*  alpha  = (float*)take((size_t)ROWS * Hh * 4);        // 512 KB
    int*    deg    = (int*)take((size_t)Nn * 4);
    int*    rowst  = (int*)take((size_t)(Nn + 1) * 4);
    int*    cursor = (int*)take((size_t)Nn * 4);
    int*    eidx   = (int*)take((size_t)Ee * 4);
    int*    srcs   = (int*)take((size_t)Ee * 4);
    __bf16* xVb    = (__bf16*)take((size_t)ROWS * HC * 2);       // 16 MB
    float*  out0f  = (float*)take((size_t)ROWS * HC * 4);        // 32 MB
    __bf16* out0bf = (__bf16*)take((size_t)ROWS * HC * 2);       // 16 MB
    __bf16* h_bf   = (__bf16*)take((size_t)ROWS * HC * 2);       // 16 MB
    float*  rff    = (float*)take((size_t)ROWS * HC * 4);        // 32 MB

    // --- precompute / conversions ---
    int nx = ROWS * Dd;
    k_f32_to_bf16<<<(nx + 255) / 256, 256, 0, stream>>>(x, x_bf, nx);
    k_f32_to_bf16<<<(HC * Dd + 255) / 256, 256, 0, stream>>>(W_V, wv_bf, HC * Dd);
    k_f32_to_bf16<<<(HC * HC + 255) / 256, 256, 0, stream>>>(W1, w1_bf, HC * HC);
    k_f32_to_bf16<<<(HC * HC + 255) / 256, 256, 0, stream>>>(W2, w2_bf, HC * HC);
    k_watt<<<(Hh * Dd + 255) / 256, 256, 0, stream>>>(W_K, b_K, att_r, wA, ab);
    k_alpha<<<(ROWS * Hh + 255) / 256, 256, 0, stream>>>(x, wA, ab, alpha);

    // --- CSR by destination (deterministic after per-bucket sort) ---
    hipMemsetAsync(deg, 0, (size_t)Nn * 4, stream);
    k_count<<<(Ee + 255) / 256, 256, 0, stream>>>(dst, deg);
    k_scan<<<1, 1024, 0, stream>>>(deg, rowst, cursor);
    k_scatter<<<(Ee + 255) / 256, 256, 0, stream>>>(dst, cursor, eidx);
    k_sortbuckets<<<(Nn + 255) / 256, 256, 0, stream>>>(rowst, eidx, src, srcs);

    // --- xV = x @ W_V^T + b_V  (bf16 WMMA, f32 accum, bf16 out for gather) ---
    k_gemm_bf16<<<ROWS / 16, 128, 0, stream>>>(x_bf, wv_bf, b_V, nullptr, xVb, ROWS, 0);

    // --- fused gather softmax-agg + residual + LN0 ---
    k_aggln<<<ROWS, 256, 0, stream>>>(alpha, xVb, rowst, srcs, att_r, ln0_g, ln0_b,
                                      out0f, out0bf);

    // --- FFN: h = relu(out0 @ W1^T + b1) [bf16]; rff = h @ W2^T + b2 ---
    k_gemm_bf16<<<ROWS / 16, 128, 0, stream>>>(out0bf, w1_bf, b1, nullptr, h_bf, ROWS, 1);
    k_gemm_bf16<<<ROWS / 16, 128, 0, stream>>>(h_bf, w2_bf, b2, rff, nullptr, ROWS, 0);

    // --- final LN ---
    k_ln1<<<ROWS, 256, 0, stream>>>(out0f, rff, ln1_g, ln1_b, out);

    (void)in_sizes; (void)n_in; (void)out_size; (void)ws_size;
}
